// A2C_loss_64518998720812
// MI455X (gfx1250) — matrix operand order repaired and checked by reading, not compiled, not run on hardware
//
#include <hip/hip_runtime.h>
#include <hip/hip_bf16.h>

// ---------------------------------------------------------------------------
// Proxy-anchor style loss, fused:
//   1) normalize rows of inst_embed / inst_proxy -> bf16 (workspace)
//   2) fused bf16 WMMA GEMM (xn @ yn^T) + masked per-row loss reduction
//   3) per-row divide + global mean
// MI455X reasoning: GEMM is 2*8192*8192*256 ~= 34 GFLOP; fused inputs are
// ~25 MB (L2-resident, 192MB L2) -> compute bound -> use the dense bf16 WMMA
// path (v_wmma_f32_16x16x32_bf16, f32 accumulate). Never materialize the
// 256MB dist matrix.
// ---------------------------------------------------------------------------

typedef __attribute__((ext_vector_type(16))) __bf16 v16bf;
typedef __attribute__((ext_vector_type(8)))  float  v8f;

#define N_INST   8192
#define N_PROXY  8192
#define DIM      256
#define NCLS     1000
#define CSPLIT   8                       // column chunks per row-tile
#define ROW_TILES (N_INST / 16)
#define COLS_PER_CHUNK (N_PROXY / CSPLIT)        // 1024
#define COL_TILES (COLS_PER_CHUNK / 16)          // 64

__device__ __forceinline__ unsigned short f32_to_bf16_rne(float f) {
    unsigned int u = __float_as_uint(f);
    unsigned int r = (u + 0x7FFFu + ((u >> 16) & 1u)) >> 16;
    return (unsigned short)r;
}

// ---------------------------------------------------------------- zero scratch
__global__ void zero_kernel(float* __restrict__ p, int n) {
    int i = blockIdx.x * blockDim.x + threadIdx.x;
    if (i < n) p[i] = 0.0f;
}

// -------------------------------------------------- row-normalize -> bf16 rows
// one wave32 per row; lane handles 8 consecutive floats (2x float4)
__global__ void normalize_kernel(const float* __restrict__ x,
                                 const float* __restrict__ y,
                                 unsigned short* __restrict__ xn,
                                 unsigned short* __restrict__ yn) {
    int wave = (blockIdx.x * blockDim.x + threadIdx.x) >> 5;
    int lane = threadIdx.x & 31;
    const float* src;
    unsigned short* dst;
    if (wave < N_INST) { src = x + (size_t)wave * DIM;           dst = xn + (size_t)wave * DIM; }
    else               { src = y + (size_t)(wave - N_INST) * DIM; dst = yn + (size_t)(wave - N_INST) * DIM; }

    const float4* s4 = (const float4*)(src + lane * 8);
    float4 a = s4[0];
    float4 b = s4[1];
    float ss = a.x*a.x + a.y*a.y + a.z*a.z + a.w*a.w
             + b.x*b.x + b.y*b.y + b.z*b.z + b.w*b.w;
    #pragma unroll
    for (int m = 1; m < 32; m <<= 1) ss += __shfl_xor(ss, m, 32);

    float nrm = fmaxf(sqrtf(ss), 1e-8f);
    float inv = 1.0f / nrm;

    float v[8] = {a.x*inv, a.y*inv, a.z*inv, a.w*inv,
                  b.x*inv, b.y*inv, b.z*inv, b.w*inv};
    uint4 packed;
    packed.x = (unsigned)f32_to_bf16_rne(v[0]) | ((unsigned)f32_to_bf16_rne(v[1]) << 16);
    packed.y = (unsigned)f32_to_bf16_rne(v[2]) | ((unsigned)f32_to_bf16_rne(v[3]) << 16);
    packed.z = (unsigned)f32_to_bf16_rne(v[4]) | ((unsigned)f32_to_bf16_rne(v[5]) << 16);
    packed.w = (unsigned)f32_to_bf16_rne(v[6]) | ((unsigned)f32_to_bf16_rne(v[7]) << 16);
    *(uint4*)(dst + lane * 8) = packed;
}

// ------------------------------------------------ fused WMMA GEMM + loss epilogue
// one wave per block; wave = (row_tile, chunk). 16 rows x 1024 cols per wave.
union V16 { v16bf v; uint4 q[2]; };

__global__ __launch_bounds__(32)
void loss_main_kernel(const unsigned short* __restrict__ xn,
                      const unsigned short* __restrict__ yn,
                      const int* __restrict__ labels,
                      const int* __restrict__ labels_proxy,
                      const int* __restrict__ real_list,
                      const int* __restrict__ is_real_p,
                      const float* __restrict__ att,
                      float* __restrict__ rowAp, float* __restrict__ rowPosC,
                      float* __restrict__ rowAn, float* __restrict__ rowHgC) {
    const int w        = blockIdx.x;
    const int row_tile = w / CSPLIT;
    const int chunk    = w % CSPLIT;
    const int i0       = row_tile * 16;
    const int lane     = threadIdx.x;
    const int half     = lane >> 4;          // 0: lanes 0-15, 1: lanes 16-31
    const int c        = lane & 15;
    const int is_real  = is_real_p[0];

    // ---- preload A fragments for all 8 K-steps (ISA 16-bit A 16x32 layout):
    // lane holds row M = lane&15; lanes 0-15 K={0..7,16..23}, lanes 16-31 K={8..15,24..31}
    const unsigned short* abase = xn + (size_t)(i0 + c) * DIM;
    V16 afrag[8];
    #pragma unroll
    for (int s = 0; s < 8; ++s) {
        afrag[s].q[0] = *(const uint4*)(abase + 32 * s + 8 * half);
        afrag[s].q[1] = *(const uint4*)(abase + 32 * s + 16 + 8 * half);
    }

    // ---- C/D layout rows for this lane: M = v + 8*half  (v = 0..7)
    const int rb = i0 + 8 * half;
    int li[8];
    const float* attrow[8];
    #pragma unroll
    for (int v = 0; v < 8; ++v) {
        li[v]     = labels[rb + v];
        attrow[v] = att + (size_t)li[v] * NCLS;
    }

    float apS[8], posC[8], anS[8], hgC[8];
    #pragma unroll
    for (int v = 0; v < 8; ++v) { apS[v] = 0.f; posC[v] = 0.f; anS[v] = 0.f; hgC[v] = 0.f; }

    const int j0base = chunk * COLS_PER_CHUNK;
    for (int t = 0; t < COL_TILES; ++t) {
        const int jc = j0base + t * 16 + c;
        // B layout (32x16 bf16): lane holds column N = lane&15, same K order as A:
        // lanes 0-15 K={0..7,16..23}, lanes 16-31 K={8..15,24..31}
        const unsigned short* bbase = yn + (size_t)jc * DIM + 8 * half;

        v8f acc = {};
        #pragma unroll
        for (int s = 0; s < 8; ++s) {
            V16 bfrag;
            bfrag.q[0] = *(const uint4*)(bbase + 32 * s);
            bfrag.q[1] = *(const uint4*)(bbase + 32 * s + 16);
            acc = __builtin_amdgcn_wmma_f32_16x16x32_bf16(
                false, afrag[s].v, false, bfrag.v, (short)0, acc, false, false);
        }

        const int lj = labels_proxy[jc];
        const int rl = real_list[jc];
        const float realok = (is_real == 0 || rl != 0) ? 1.0f : 0.0f;

        #pragma unroll
        for (int v = 0; v < 8; ++v) {
            const float dist = 1.0f - acc[v];
            const float a_th = attrow[v][lj] * 0.5f + 0.4f;   // BIAS
            const bool  eq   = (li[v] == lj);
            const float pm   = eq ? realok : 0.0f;            // pos & real
            const float nm   = eq ? 0.0f   : realok;          // neg & real
            apS[v]  += pm * fmaxf(dist - 0.05f, 0.0f);
            posC[v] += pm;
            const float hm = (dist < a_th) ? nm : 0.0f;       // hinge
            anS[v]  += hm * (a_th - dist);
            hgC[v]  += hm;
        }
    }

    // ---- reduce over the 16 columns held by each 16-lane group
    #pragma unroll
    for (int m = 1; m < 16; m <<= 1) {
        #pragma unroll
        for (int v = 0; v < 8; ++v) {
            apS[v]  += __shfl_xor(apS[v],  m, 32);
            posC[v] += __shfl_xor(posC[v], m, 32);
            anS[v]  += __shfl_xor(anS[v],  m, 32);
            hgC[v]  += __shfl_xor(hgC[v],  m, 32);
        }
    }
    if (c == 0) {
        #pragma unroll
        for (int v = 0; v < 8; ++v) {
            atomicAdd(&rowAp[rb + v],   apS[v]);
            atomicAdd(&rowPosC[rb + v], posC[v]);
            atomicAdd(&rowAn[rb + v],   anS[v]);
            atomicAdd(&rowHgC[rb + v],  hgC[v]);
        }
    }
}

// ------------------------------------------------------- per-row loss + mean
__global__ void finalize_kernel(const float* __restrict__ rowAp,
                                const float* __restrict__ rowPosC,
                                const float* __restrict__ rowAn,
                                const float* __restrict__ rowHgC,
                                float* __restrict__ out) {
    __shared__ float sdata[256];
    const int tid = threadIdx.x;
    float s = 0.0f;
    for (int i = tid; i < N_INST; i += 256) {
        const float lap = rowAp[i] / (rowPosC[i] + 1e-5f);
        const float lan = rowAn[i] / (rowHgC[i]  + 1e-5f);
        s += lap + lan;
    }
    sdata[tid] = s;
    __syncthreads();
    for (int st = 128; st > 0; st >>= 1) {
        if (tid < st) sdata[tid] += sdata[tid + st];
        __syncthreads();
    }
    if (tid == 0) out[0] = sdata[0] / (float)N_INST;
}

// ---------------------------------------------------------------------------
extern "C" void kernel_launch(void* const* d_in, const int* in_sizes, int n_in,
                              void* d_out, int out_size, void* d_ws, size_t ws_size,
                              hipStream_t stream) {
    const float* inst_embed   = (const float*)d_in[0];
    const int*   labels       = (const int*)  d_in[1];
    const float* inst_proxy   = (const float*)d_in[2];
    const int*   labels_proxy = (const int*)  d_in[3];
    // d_in[4]=margin, d_in[5]=alpha : unused by the reference math
    const int*   real_list    = (const int*)  d_in[6];
    const int*   is_real      = (const int*)  d_in[7];
    const float* att_distance = (const float*)d_in[8];

    // workspace layout: xn(bf16) | yn(bf16) | 4 per-row f32 accumulators
    unsigned short* xn = (unsigned short*)d_ws;
    unsigned short* yn = xn + (size_t)N_INST * DIM;
    float* rowAp   = (float*)(yn + (size_t)N_PROXY * DIM);
    float* rowPosC = rowAp   + N_INST;
    float* rowAn   = rowPosC + N_INST;
    float* rowHgC  = rowAn   + N_INST;

    zero_kernel<<<(4 * N_INST + 255) / 256, 256, 0, stream>>>(rowAp, 4 * N_INST);

    // 16384 rows, one wave each, 8 waves per 256-thread block
    normalize_kernel<<<(N_INST + N_PROXY) / 8, 256, 0, stream>>>(
        inst_embed, inst_proxy, xn, yn);

    loss_main_kernel<<<ROW_TILES * CSPLIT, 32, 0, stream>>>(
        xn, yn, labels, labels_proxy, real_list, is_real, att_distance,
        rowAp, rowPosC, rowAn, rowHgC);

    finalize_kernel<<<1, 256, 0, stream>>>(rowAp, rowPosC, rowAn, rowHgC,
                                           (float*)d_out);
}
